// BigramSeg_7438883357468
// MI455X (gfx1250) — matrix operand order repaired and checked by reading, not compiled, not run on hardware
//
#include <hip/hip_runtime.h>

// ---------------------------------------------------------------------------
// Problem constants (from reference): B=64, T=512, D=768, HID=512, C=32
// ---------------------------------------------------------------------------
constexpr int Bsz  = 64;
constexpr int Tlen = 512;
constexpr int Dft  = 768;
constexpr int HID  = 512;
constexpr int Ccls = 32;
constexpr int INsz = Dft + Ccls;   // 800; note Dft = 24*32, so K-step 24 = one-hot
constexpr int KSTEP = 32;          // WMMA bf16 K depth
constexpr int NTHREADS = 512;      // 16 waves -> 4 waves per SIMD32 for latency hiding

typedef __bf16 bf16_t;
typedef bf16_t v16bf __attribute__((ext_vector_type(16)));
typedef bf16_t v8bf  __attribute__((ext_vector_type(8)));
typedef bf16_t v4bf  __attribute__((ext_vector_type(4)));
typedef float  v8f   __attribute__((ext_vector_type(8)));

union FragBF16 { v16bf v; v8bf h[2]; };

// ---------------------------------------------------------------------------
// One-time weight conversion f32 -> bf16 (weights then live in 192MB L2).
// ---------------------------------------------------------------------------
__global__ void cvt_weights_kernel(const float* __restrict__ W1,
                                   const float* __restrict__ W2,
                                   bf16_t* __restrict__ W1b,
                                   bf16_t* __restrict__ W2b)
{
    int i = blockIdx.x * blockDim.x + threadIdx.x;
    if (i < HID * INsz) W1b[i] = (bf16_t)W1[i];
    if (i < Ccls * HID) W2b[i] = (bf16_t)W2[i];
}

// One-time X conversion f32 -> bf16 (enables raw async LDS copies in main loop)
__global__ void cvt_x_kernel(const float* __restrict__ X, bf16_t* __restrict__ Xb)
{
    long i = (long)blockIdx.x * blockDim.x + threadIdx.x;   // float4 index
    if (i < (long)Bsz * Tlen * Dft / 4) {
        float4 x4 = *(const float4*)(X + i * 4);
        v4bf h4 = { (bf16_t)x4.x, (bf16_t)x4.y, (bf16_t)x4.z, (bf16_t)x4.w };
        *(v4bf*)(Xb + i * 4) = h4;
    }
}

// ---------------------------------------------------------------------------
// Autoregressive kernel: one block per 16-row batch tile, 16 waves (wave32).
// ASYNCX=true : x tiles are double-buffered into LDS with
//               global_load_async_to_lds_b128 (ASYNCcnt), copy for t+1 hidden
//               behind the WMMA work of step t.
// ASYNCX=false: fallback, per-step f32->bf16 conversion into LDS.
// ---------------------------------------------------------------------------
template <bool ASYNCX>
__global__ __launch_bounds__(NTHREADS, 1)
void bigram_ar_kernel(const float* __restrict__ X,      // [B,T,D] f32
                      const bf16_t* __restrict__ Xb,    // [B,T,D] bf16 (ASYNCX)
                      const bf16_t* __restrict__ W1b,   // [HID,IN] bf16
                      const float* __restrict__ b1,     // [HID]
                      const bf16_t* __restrict__ W2b,   // [C,HID] bf16
                      const float* __restrict__ b2,     // [C]
                      float* __restrict__ out)          // logits|probs|preds|one_hots
{
    __shared__ bf16_t sX[2][16][Dft];    // double-buffered x_t tile (24 KB each)
    __shared__ bf16_t sOH[16][Ccls];     // one-hot feedback columns (K-step 24)
    __shared__ bf16_t sH[16][HID];       // tanh(h) tile
    __shared__ float  sLogit[16][Ccls];
    __shared__ float  sInv[16];
    __shared__ float  sMax[16];
    __shared__ int    sPred[16];

    const int tid   = threadIdx.x;
    const int wave  = tid >> 5;
    const int lane  = tid & 31;
    const int lhalf = lane & 15;
    const bool hi   = (lane >= 16);
    const int b0    = blockIdx.x * 16;

    const long BTC = (long)Bsz * Tlen * Ccls;
    float* outLogits = out;
    float* outProbs  = out + BTC;
    float* outPreds  = out + 2 * BTC;
    float* outOneh   = out + 2 * BTC + (long)Bsz * Tlen;

    // issue async copy of one 16 x 768 bf16 tile (16B per lane-op)
    auto issue_tile_copy = [&](int t, int nb) {
        for (int idx = tid; idx < 16 * (Dft / 8); idx += NTHREADS) {
            int r = idx / (Dft / 8);
            int c = (idx % (Dft / 8)) * 8;
            const bf16_t* g = Xb + ((long)(b0 + r) * Tlen + t) * Dft + c;
            unsigned lds = (unsigned)(uintptr_t)(void*)&sX[nb][r][c];
            asm volatile("global_load_async_to_lds_b128 %0, %1, off"
                         :: "v"(lds), "v"(g) : "memory");
        }
    };

    // t=0 feedback one-hot is all zeros
    for (int i = tid; i < 16 * Ccls; i += NTHREADS)
        sOH[i >> 5][i & 31] = (bf16_t)0.0f;

    if (ASYNCX) issue_tile_copy(0, 0);   // prologue: stage tile for t=0

    for (int t = 0; t < Tlen; ++t) {
        const int buf = t & 1;

        // ---- Phase 1: make x tile for step t visible ----------------------
        if (ASYNCX) {
            asm volatile("s_wait_asynccnt 0x0" ::: "memory");  // own copies done
            __syncthreads();                                    // everyone's done
            if (t + 1 < Tlen) issue_tile_copy(t + 1, buf ^ 1);  // hide behind GEMMs
        } else {
            for (int idx = tid; idx < 16 * (Dft / 4); idx += NTHREADS) {
                int r = idx / (Dft / 4);
                int v = idx % (Dft / 4);
                float4 x4 = *(const float4*)(X + ((long)(b0 + r) * Tlen + t) * Dft + v * 4);
                v4bf h4 = { (bf16_t)x4.x, (bf16_t)x4.y, (bf16_t)x4.z, (bf16_t)x4.w };
                *(v4bf*)&sX[buf][r][v * 4] = h4;
            }
            if (t + 1 < Tlen)
                __builtin_prefetch(X + ((long)(b0 + lhalf) * Tlen + (t + 1)) * Dft, 0, 1);
            __syncthreads();
        }

        // ---- Phase 2: GEMM1  h = [x|oh][16x800] @ W1^T -> [16x512] --------
        // wave w owns output columns [w*32, w*32+32): 2 N-tiles of 16.
        v8f acc[2];
        const int n0 = wave * 32;
#pragma unroll
        for (int i = 0; i < 2; ++i) {
            float bv = b1[n0 + i * 16 + lhalf];
#pragma unroll
            for (int r = 0; r < 8; ++r) acc[i][r] = bv;
        }

#pragma unroll
        for (int kk = 0; kk < INsz / KSTEP; ++kk) {
            // A fragment (16x32 bf16): lanes 0-15 -> K 0..7 & 16..23 of row m,
            // lanes 16-31 -> K 8..15 & 24..31. kk==24 reads the one-hot block.
            const bf16_t* arow = (kk < Dft / KSTEP) ? &sX[buf][lhalf][kk * KSTEP]
                                                    : &sOH[lhalf][0];
            const int a0 = hi ? 8 : 0;
            FragBF16 A;
            A.h[0] = *(const v8bf*)(arow + a0);
            A.h[1] = *(const v8bf*)(arow + a0 + 16);

            const int kB = kk * KSTEP + (hi ? 8 : 0);
#pragma unroll
            for (int i = 0; i < 2; ++i) {
                // B = W1^T: column n of B is row n of W1 -> contiguous in k.
                const bf16_t* wrow = W1b + (long)(n0 + i * 16 + lhalf) * INsz + kB;
                FragBF16 Bm;
                Bm.h[0] = *(const v8bf*)(wrow);
                Bm.h[1] = *(const v8bf*)(wrow + 16);
                acc[i] = __builtin_amdgcn_wmma_f32_16x16x32_bf16(
                    false, A.v, false, Bm.v, (short)0, acc[i], false, false);
            }
        }

        // ---- Phase 3: tanh -> sH (bf16) -----------------------------------
#pragma unroll
        for (int i = 0; i < 2; ++i) {
            int n = n0 + i * 16 + lhalf;
#pragma unroll
            for (int r = 0; r < 8; ++r) {
                int m = hi ? (r + 8) : r;
                sH[m][n] = (bf16_t)tanhf(acc[i][r]);
            }
        }
        __syncthreads();

        // ---- Phase 4: GEMM2  logit = h[16x512] @ W2^T -> [16x32] ----------
        if (wave < 2) {
            const int nc = wave * 16 + lhalf;
            v8f acc2;
            float bv = b2[nc];
#pragma unroll
            for (int r = 0; r < 8; ++r) acc2[r] = bv;

#pragma unroll
            for (int kk = 0; kk < HID / KSTEP; ++kk) {
                const int kA = kk * KSTEP + (hi ? 8 : 0);
                FragBF16 A;
                A.h[0] = *(const v8bf*)&sH[lhalf][kA];
                A.h[1] = *(const v8bf*)&sH[lhalf][kA + 16];

                const int kB = kk * KSTEP + (hi ? 8 : 0);
                const bf16_t* wrow = W2b + (long)nc * HID + kB;
                FragBF16 Bm;
                Bm.h[0] = *(const v8bf*)(wrow);
                Bm.h[1] = *(const v8bf*)(wrow + 16);
                acc2 = __builtin_amdgcn_wmma_f32_16x16x32_bf16(
                    false, A.v, false, Bm.v, (short)0, acc2, false, false);
            }
#pragma unroll
            for (int r = 0; r < 8; ++r) {
                int m = hi ? (r + 8) : r;
                sLogit[m][nc] = acc2[r];
            }
        }
        __syncthreads();

        // ---- Phase 5: softmax stats / argmax / one-hot feedback -----------
        if (tid < 16) {
            float mx = sLogit[tid][0];
            int am = 0;
            for (int c = 1; c < Ccls; ++c) {
                float v = sLogit[tid][c];
                if (v > mx) { mx = v; am = c; }
            }
            float s = 0.0f;
            for (int c = 0; c < Ccls; ++c) s += __expf(sLogit[tid][c] - mx);
            sMax[tid] = mx;
            sInv[tid] = 1.0f / s;
            sPred[tid] = am;
            for (int c = 0; c < Ccls; ++c)
                sOH[tid][c] = (bf16_t)((c == am) ? 1.0f : 0.0f);
        }
        __syncthreads();

        // ---- Phase 6: write outputs ---------------------------------------
        for (int idx = tid; idx < 16 * Ccls; idx += NTHREADS) {
            int r = idx >> 5, c = idx & 31;
            long g = ((long)(b0 + r) * Tlen + t) * Ccls + c;
            float lg = sLogit[r][c];
            outLogits[g] = lg;
            outProbs[g]  = __expf(lg - sMax[r]) * sInv[r];
            outOneh[g]   = (c == sPred[r]) ? 1.0f : 0.0f;
        }
        if (tid < 16)
            outPreds[(long)(b0 + tid) * Tlen + t] = (float)sPred[tid];
        // (top-of-next-step barrier covers remaining hazards)
    }
}

// ---------------------------------------------------------------------------
extern "C" void kernel_launch(void* const* d_in, const int* in_sizes, int n_in,
                              void* d_out, int out_size, void* d_ws, size_t ws_size,
                              hipStream_t stream) {
    const float* X  = (const float*)d_in[0];
    // d_in[1] = lengths (unused by reference outputs)
    const float* W1 = (const float*)d_in[2];
    const float* b1 = (const float*)d_in[3];
    const float* W2 = (const float*)d_in[4];
    const float* b2 = (const float*)d_in[5];
    float* out = (float*)d_out;

    const size_t w1bytes = (size_t)HID * INsz * sizeof(bf16_t);   // 819,200
    const size_t w2bytes = (size_t)Ccls * HID * sizeof(bf16_t);   //  32,768
    const size_t xbytes  = (size_t)Bsz * Tlen * Dft * sizeof(bf16_t); // 50,331,648

    bf16_t* W1b = (bf16_t*)d_ws;
    bf16_t* W2b = (bf16_t*)((char*)d_ws + w1bytes);
    bf16_t* Xb  = (bf16_t*)((char*)d_ws + w1bytes + w2bytes);     // 256B aligned

    cvt_weights_kernel<<<(HID * INsz + 255) / 256, 256, 0, stream>>>(W1, W2, W1b, W2b);

    const bool useAsync = ws_size >= (w1bytes + w2bytes + xbytes);
    if (useAsync) {
        long nvec4 = (long)Bsz * Tlen * Dft / 4;
        cvt_x_kernel<<<(int)((nvec4 + 255) / 256), 256, 0, stream>>>(X, Xb);
        bigram_ar_kernel<true><<<Bsz / 16, NTHREADS, 0, stream>>>(
            X, Xb, W1b, b1, W2b, b2, out);
    } else {
        bigram_ar_kernel<false><<<Bsz / 16, NTHREADS, 0, stream>>>(
            X, Xb, W1b, b1, W2b, b2, out);
    }
}
